// Seq2Seq_76733885710449
// MI455X (gfx1250) — compile-verified, matrix operand order checked by reading, not run on hardware
//
#include <hip/hip_runtime.h>
#include <hip/hip_bf16.h>

typedef __attribute__((ext_vector_type(16))) __bf16 v16bf;
typedef __attribute__((ext_vector_type(8)))  __bf16 v8bf;
typedef __attribute__((ext_vector_type(8)))  float  v8f;

constexpr int S  = 50;
constexpr int T  = 50;
constexpr int B  = 64;
constexpr int E  = 256;
constexpr int H  = 512;
constexpr int VT = 10000;

__device__ __forceinline__ unsigned short f2bf(float f) {
    union { float f; unsigned u; } v; v.f = f;
    unsigned r = v.u + 0x7FFFu + ((v.u >> 16) & 1u);   // round-to-nearest-even
    return (unsigned short)(r >> 16);
}

// ---------------------------------------------------------------------------
// WMMA GEMM: C[M,N] = A[M,K](bf16) * W[N,K]^T(bf16) + bias[N], optional ReLU.
// One wave32 per 64x16 C strip (4 M-tiles register-blocked): the B (weight)
// tile is loaded ONCE per 32-wide K-chunk and reused by 4 WMMAs, quartering
// weight-side L2 bandwidth — the binding resource since M is only 64.
// Per-lane operand layout per CDNA5 ISA 7.12.2 (16-bit A-matrix 16x32):
//   lane L holds row L%16; K-half kh=L/16; group g in {0,1}:
//   halves [g*16 + kh*8, +8) are contiguous -> one b128 load per group.
// Requires M % 64 == 0, N % 16 == 0, K % 32 == 0 (true for all call sites).
// ---------------------------------------------------------------------------
__global__ void k_gemm_bf16(const unsigned short* __restrict__ A,
                            const unsigned short* __restrict__ W,
                            const float* __restrict__ bias,
                            float* __restrict__ C,
                            int M, int N, int K, int relu) {
    const int lane = threadIdx.x;          // 0..31
    const int tn   = blockIdx.x;           // N/16 tiles
    const int tm   = blockIdx.y;           // M/64 strips
    const int r16  = lane & 15;
    const int kh   = lane >> 4;            // 0 or 1

    const unsigned short* Abase = A + (size_t)(tm * 64 + r16) * K + kh * 8;
    const unsigned short* Wp    = W + (size_t)(tn * 16 + r16) * K + kh * 8;

    v8f acc0 = {}, acc1 = {}, acc2 = {}, acc3 = {};
    for (int k0 = 0; k0 < K; k0 += 32) {
        // prefetch weight stream one chunk-group ahead (linear in K)
        __builtin_prefetch(Wp + k0 + 256, 0, 1);

        v8bf b0 = *reinterpret_cast<const v8bf*>(Wp + k0);
        v8bf b1 = *reinterpret_cast<const v8bf*>(Wp + k0 + 16);
        v16bf b = __builtin_shufflevector(b0, b1, 0,1,2,3,4,5,6,7,8,9,10,11,12,13,14,15);

        v8bf a00 = *reinterpret_cast<const v8bf*>(Abase + (size_t)0 * 16 * K + k0);
        v8bf a01 = *reinterpret_cast<const v8bf*>(Abase + (size_t)0 * 16 * K + k0 + 16);
        v8bf a10 = *reinterpret_cast<const v8bf*>(Abase + (size_t)1 * 16 * K + k0);
        v8bf a11 = *reinterpret_cast<const v8bf*>(Abase + (size_t)1 * 16 * K + k0 + 16);
        v8bf a20 = *reinterpret_cast<const v8bf*>(Abase + (size_t)2 * 16 * K + k0);
        v8bf a21 = *reinterpret_cast<const v8bf*>(Abase + (size_t)2 * 16 * K + k0 + 16);
        v8bf a30 = *reinterpret_cast<const v8bf*>(Abase + (size_t)3 * 16 * K + k0);
        v8bf a31 = *reinterpret_cast<const v8bf*>(Abase + (size_t)3 * 16 * K + k0 + 16);

        v16bf a0 = __builtin_shufflevector(a00, a01, 0,1,2,3,4,5,6,7,8,9,10,11,12,13,14,15);
        v16bf a1 = __builtin_shufflevector(a10, a11, 0,1,2,3,4,5,6,7,8,9,10,11,12,13,14,15);
        v16bf a2 = __builtin_shufflevector(a20, a21, 0,1,2,3,4,5,6,7,8,9,10,11,12,13,14,15);
        v16bf a3 = __builtin_shufflevector(a30, a31, 0,1,2,3,4,5,6,7,8,9,10,11,12,13,14,15);

        acc0 = __builtin_amdgcn_wmma_f32_16x16x32_bf16(false, a0, false, b, (short)0, acc0, false, false);
        acc1 = __builtin_amdgcn_wmma_f32_16x16x32_bf16(false, a1, false, b, (short)0, acc1, false, false);
        acc2 = __builtin_amdgcn_wmma_f32_16x16x32_bf16(false, a2, false, b, (short)0, acc2, false, false);
        acc3 = __builtin_amdgcn_wmma_f32_16x16x32_bf16(false, a3, false, b, (short)0, acc3, false, false);
    }

    // C/D layout: VGPR i -> m = mtile*16 + i + 8*kh, n = tn*16 + r16
    const int n  = tn * 16 + r16;
    const float bv = bias ? bias[n] : 0.0f;
    v8f accs[4] = {acc0, acc1, acc2, acc3};
#pragma unroll
    for (int mt = 0; mt < 4; ++mt) {
        const int mbase = tm * 64 + mt * 16 + kh * 8;
#pragma unroll
        for (int i = 0; i < 8; ++i) {
            float v = accs[mt][i] + bv;
            if (relu) v = fmaxf(v, 0.0f);
            C[(size_t)(mbase + i) * N + n] = v;
        }
    }
}

// Gather embedding rows -> bf16
__global__ void k_embed(const int* __restrict__ tok, const float* __restrict__ table,
                        unsigned short* __restrict__ out, int rows, int cols) {
    int idx = blockIdx.x * 256 + threadIdx.x;
    if (idx >= rows * cols) return;
    int r = idx / cols, c = idx - r * cols;
    out[idx] = f2bf(table[(size_t)tok[r] * cols + c]);
}

// fp32 -> bf16 with row strides (also used for plain contiguous casts)
__global__ void k_cast_strided(const float* __restrict__ src, int srcStride,
                               unsigned short* __restrict__ dst, int dstStride,
                               int dstOff, int rows, int cols) {
    int idx = blockIdx.x * 256 + threadIdx.x;
    if (idx >= rows * cols) return;
    int r = idx / cols, c = idx - r * cols;
    dst[(size_t)r * dstStride + dstOff + c] = f2bf(src[(size_t)r * srcStride + c]);
}

__global__ void k_copy_bf16_strided(const unsigned short* __restrict__ src, int srcStride,
                                    unsigned short* __restrict__ dst, int dstStride,
                                    int dstOff, int rows, int cols) {
    int idx = blockIdx.x * 256 + threadIdx.x;
    if (idx >= rows * cols) return;
    int r = idx / cols, c = idx - r * cols;
    dst[(size_t)r * dstStride + dstOff + c] = src[(size_t)r * srcStride + c];
}

__global__ void k_zero(float* __restrict__ p, int n) {
    int i = blockIdx.x * 256 + threadIdx.x;
    if (i < n) p[i] = 0.0f;
}

// GRU gate fusion: gi/gh are [B,3H] (r,z,n); h,hnew are [B,H].
// Optionally mirrors h_new into out2[b*stride + off + j] (encoder outputs).
__global__ void k_gru(const float* __restrict__ gi, const float* __restrict__ gh,
                      const float* __restrict__ h, float* __restrict__ hnew,
                      float* __restrict__ out2, int out2_stride, int out2_off) {
    int idx = blockIdx.x * 256 + threadIdx.x;
    if (idx >= B * H) return;
    int b = idx / H, j = idx - b * H;
    const float* gib = gi + (size_t)b * 3 * H;
    const float* ghb = gh + (size_t)b * 3 * H;
    float r = 1.0f / (1.0f + __expf(-(gib[j]         + ghb[j])));
    float z = 1.0f / (1.0f + __expf(-(gib[H + j]     + ghb[H + j])));
    float n = tanhf(gib[2 * H + j] + r * ghb[2 * H + j]);
    float hv = (1.0f - z) * n + z * h[idx];
    hnew[idx] = hv;
    if (out2) out2[(size_t)b * out2_stride + out2_off + j] = hv;
}

// scores[b,s] = sum_h relu(q[b,h] + keys[s,b,h]) * va[h];  grid(S,B), block 128
__global__ void k_attn_scores(const float* __restrict__ q, const float* __restrict__ keys,
                              const float* __restrict__ va, float* __restrict__ scores) {
    int s = blockIdx.x, b = blockIdx.y;
    __shared__ float red[128];
    const float* kp = keys + ((size_t)s * B + b) * H;
    const float* qp = q + (size_t)b * H;
    float acc = 0.0f;
    for (int h = threadIdx.x; h < H; h += 128) {
        float e = qp[h] + kp[h];
        e = e > 0.0f ? e : 0.0f;
        acc += e * va[h];
    }
    red[threadIdx.x] = acc;
    __syncthreads();
    for (int offw = 64; offw > 0; offw >>= 1) {
        if ((int)threadIdx.x < offw) red[threadIdx.x] += red[threadIdx.x + offw];
        __syncthreads();
    }
    if (threadIdx.x == 0) scores[b * S + s] = red[0];
}

// softmax over S per batch row; grid(B), block 64
__global__ void k_softmax(const float* __restrict__ scores, float* __restrict__ attn) {
    int b = blockIdx.x, t = threadIdx.x;
    __shared__ float sm[64];
    float v = (t < S) ? scores[b * S + t] : -1e30f;
    sm[t] = v;
    __syncthreads();
    for (int o = 32; o > 0; o >>= 1) { if (t < o) sm[t] = fmaxf(sm[t], sm[t + o]); __syncthreads(); }
    float mx = sm[0];
    __syncthreads();
    float e = (t < S) ? __expf(v - mx) : 0.0f;
    sm[t] = e;
    __syncthreads();
    for (int o = 32; o > 0; o >>= 1) { if (t < o) sm[t] += sm[t + o]; __syncthreads(); }
    if (t < S) attn[b * S + t] = e / sm[0];
}

// ctx[b,k] = sum_s attn[b,s] * enc_out[s,b,k], k < 2H; grid(B), block 256
__global__ void k_ctx(const float* __restrict__ attn, const float* __restrict__ enc_out,
                      float* __restrict__ ctx) {
    int b = blockIdx.x;
    for (int k = threadIdx.x; k < 2 * H; k += 256) {
        float acc = 0.0f;
        for (int s = 0; s < S; ++s)
            acc += attn[b * S + s] * enc_out[((size_t)s * B + b) * (2 * H) + k];
        ctx[(size_t)b * (2 * H) + k] = acc;
    }
}

// ---------------------------------------------------------------------------
static inline int cdiv(int a, int b) { return (a + b - 1) / b; }

extern "C" void kernel_launch(void* const* d_in, const int* in_sizes, int n_in,
                              void* d_out, int out_size, void* d_ws, size_t ws_size,
                              hipStream_t stream) {
    const int*   source   = (const int*)d_in[0];
    const int*   target   = (const int*)d_in[1];
    const float* emb_enc  = (const float*)d_in[2];
    const float* Wi_f     = (const float*)d_in[3];
    const float* Wh_f     = (const float*)d_in[4];
    const float* bi_f     = (const float*)d_in[5];
    const float* bh_f     = (const float*)d_in[6];
    const float* Wi_b     = (const float*)d_in[7];
    const float* Wh_b     = (const float*)d_in[8];
    const float* bi_b     = (const float*)d_in[9];
    const float* bh_b     = (const float*)d_in[10];
    const float* fc_enc_w = (const float*)d_in[11];
    const float* fc_enc_b = (const float*)d_in[12];
    const float* emb_dec  = (const float*)d_in[13];
    const float* Wi_d     = (const float*)d_in[14];
    const float* Wh_d     = (const float*)d_in[15];
    const float* bi_d     = (const float*)d_in[16];
    const float* bh_d     = (const float*)d_in[17];
    const float* wa       = (const float*)d_in[18];
    const float* ua       = (const float*)d_in[19];
    const float* va       = (const float*)d_in[20];
    const float* fc_w     = (const float*)d_in[21];
    const float* fc_b     = (const float*)d_in[22];
    float* out = (float*)d_out;

    // bump allocator over workspace
    char* ws = (char*)d_ws;
    size_t woff = 0;
    auto alloc = [&](size_t bytes) -> void* {
        void* p = ws + woff;
        woff += (bytes + 255) & ~(size_t)255;
        return p;
    };
    auto allocBF = [&](size_t n) { return (unsigned short*)alloc(n * 2); };
    auto allocF  = [&](size_t n) { return (float*)alloc(n * 4); };

    // bf16 weight copies
    unsigned short* wif_bf   = allocBF((size_t)3 * H * E);
    unsigned short* whf_bf   = allocBF((size_t)3 * H * H);
    unsigned short* wib_bf   = allocBF((size_t)3 * H * E);
    unsigned short* whb_bf   = allocBF((size_t)3 * H * H);
    unsigned short* fcenc_bf = allocBF((size_t)H * 2 * H);
    unsigned short* wid_bf   = allocBF((size_t)3 * H * (E + 2 * H));
    unsigned short* whd_bf   = allocBF((size_t)3 * H * H);
    unsigned short* wa_bf    = allocBF((size_t)H * H);
    unsigned short* ua_bf    = allocBF((size_t)H * 2 * H);
    unsigned short* fcw_bf   = allocBF((size_t)VT * (E + 3 * H));
    // activations
    unsigned short* src_emb  = allocBF((size_t)S * B * E);
    unsigned short* tgt_emb  = allocBF((size_t)T * B * E);
    unsigned short* eob_bf   = allocBF((size_t)S * B * 2 * H);
    unsigned short* hcat_bf  = allocBF((size_t)B * 2 * H);
    unsigned short* h_bf     = allocBF((size_t)B * H);
    unsigned short* x_bf     = allocBF((size_t)B * (E + 2 * H));
    unsigned short* fcin_bf  = allocBF((size_t)B * (E + 3 * H));
    float* gi_f    = allocF((size_t)S * B * 3 * H);
    float* gi_b    = allocF((size_t)S * B * 3 * H);
    float* gh      = allocF((size_t)B * 3 * H);
    float* h_f     = allocF((size_t)B * H);
    float* h_b     = allocF((size_t)B * H);
    float* enc_out = allocF((size_t)S * B * 2 * H);
    float* keys    = allocF((size_t)S * B * H);
    float* h_dec   = allocF((size_t)B * H);
    float* q       = allocF((size_t)B * H);
    float* scores  = allocF((size_t)B * S);
    float* attn    = allocF((size_t)B * S);
    float* ctx     = allocF((size_t)B * 2 * H);
    float* gi_d    = allocF((size_t)B * 3 * H);

    auto cast = [&](const float* s, int ss, unsigned short* d, int ds, int doff,
                    int rows, int cols) {
        k_cast_strided<<<cdiv(rows * cols, 256), 256, 0, stream>>>(s, ss, d, ds, doff, rows, cols);
    };
    auto copybf = [&](const unsigned short* s, int ss, unsigned short* d, int ds, int doff,
                      int rows, int cols) {
        k_copy_bf16_strided<<<cdiv(rows * cols, 256), 256, 0, stream>>>(s, ss, d, ds, doff, rows, cols);
    };
    auto gemm = [&](const unsigned short* A, const unsigned short* W, const float* bias,
                    float* C, int M, int N, int K, int relu) {
        dim3 g(N / 16, M / 64);           // 64-row strips, M % 64 == 0 at all call sites
        k_gemm_bf16<<<g, 32, 0, stream>>>(A, W, bias, C, M, N, K, relu);
    };

    // ---- convert weights to bf16 (deterministic each call) ----
    cast(Wi_f,     E,         wif_bf,   E,         0, 3 * H, E);
    cast(Wh_f,     H,         whf_bf,   H,         0, 3 * H, H);
    cast(Wi_b,     E,         wib_bf,   E,         0, 3 * H, E);
    cast(Wh_b,     H,         whb_bf,   H,         0, 3 * H, H);
    cast(fc_enc_w, 2 * H,     fcenc_bf, 2 * H,     0, H,     2 * H);
    cast(Wi_d,     E + 2 * H, wid_bf,   E + 2 * H, 0, 3 * H, E + 2 * H);
    cast(Wh_d,     H,         whd_bf,   H,         0, 3 * H, H);
    cast(wa,       H,         wa_bf,    H,         0, H,     H);
    cast(ua,       2 * H,     ua_bf,    2 * H,     0, H,     2 * H);
    cast(fc_w,     E + 3 * H, fcw_bf,   E + 3 * H, 0, VT,    E + 3 * H);

    // ---- embeddings ----
    k_embed<<<cdiv(S * B * E, 256), 256, 0, stream>>>(source, emb_enc, src_emb, S * B, E);
    k_embed<<<cdiv(T * B * E, 256), 256, 0, stream>>>(target, emb_dec, tgt_emb, T * B, E);

    // ---- encoder: batched input-side GEMMs over all timesteps ----
    gemm(src_emb, wif_bf, bi_f, gi_f, S * B, 3 * H, E, 0);
    gemm(src_emb, wib_bf, bi_b, gi_b, S * B, 3 * H, E, 0);

    k_zero<<<cdiv(B * H, 256), 256, 0, stream>>>(h_f, B * H);
    k_zero<<<cdiv(B * H, 256), 256, 0, stream>>>(h_b, B * H);

    for (int t = 0; t < S; ++t) {
        // forward direction (consumes emb[t], writes enc_out[t][:, 0:H])
        cast(h_f, H, h_bf, H, 0, B, H);
        gemm(h_bf, whf_bf, bh_f, gh, B, 3 * H, H, 0);
        k_gru<<<cdiv(B * H, 256), 256, 0, stream>>>(
            gi_f + (size_t)t * B * 3 * H, gh, h_f, h_f,
            enc_out + (size_t)t * B * 2 * H, 2 * H, 0);
        // backward direction (consumes emb[S-1-t], writes enc_out[S-1-t][:, H:2H])
        int tb = S - 1 - t;
        cast(h_b, H, h_bf, H, 0, B, H);
        gemm(h_bf, whb_bf, bh_b, gh, B, 3 * H, H, 0);
        k_gru<<<cdiv(B * H, 256), 256, 0, stream>>>(
            gi_b + (size_t)tb * B * 3 * H, gh, h_b, h_b,
            enc_out + (size_t)tb * B * 2 * H, 2 * H, H);
    }

    // hidden = relu([h_f_last, h_b_last] @ fc_enc_w^T + b)
    cast(h_f, H, hcat_bf, 2 * H, 0, B, H);
    cast(h_b, H, hcat_bf, 2 * H, H, B, H);
    gemm(hcat_bf, fcenc_bf, fc_enc_b, h_dec, B, H, 2 * H, 1);

    // keys = enc_out @ ua^T  (batched over S*B rows)
    cast(enc_out, 2 * H, eob_bf, 2 * H, 0, S * B, 2 * H);
    gemm(eob_bf, ua_bf, nullptr, keys, S * B, H, 2 * H, 0);

    // ---- decoder ----
    k_zero<<<cdiv(B * VT, 256), 256, 0, stream>>>(out, B * VT);   // outputs[0] = 0

    for (int t = 0; t < T - 1; ++t) {
        cast(h_dec, H, h_bf, H, 0, B, H);
        // q = h @ wa^T
        gemm(h_bf, wa_bf, nullptr, q, B, H, H, 0);
        // attention
        k_attn_scores<<<dim3(S, B), 128, 0, stream>>>(q, keys, va, scores);
        k_softmax<<<B, 64, 0, stream>>>(scores, attn);
        k_ctx<<<B, 256, 0, stream>>>(attn, enc_out, ctx);
        // x = [emb_dec[target[t]], ctx]  (bf16)
        copybf(tgt_emb + (size_t)t * B * E, E, x_bf, E + 2 * H, 0, B, E);
        cast(ctx, 2 * H, x_bf, E + 2 * H, E, B, 2 * H);
        // GRU cell
        gemm(x_bf, wid_bf, bi_d, gi_d, B, 3 * H, E + 2 * H, 0);
        gemm(h_bf, whd_bf, bh_d, gh, B, 3 * H, H, 0);
        k_gru<<<cdiv(B * H, 256), 256, 0, stream>>>(gi_d, gh, h_dec, h_dec,
                                                    nullptr, 0, 0);
        // pred = [e, h2, ctx] @ fc_w^T + fc_b  -> out[t+1]
        copybf(tgt_emb + (size_t)t * B * E, E, fcin_bf, E + 3 * H, 0, B, E);
        cast(h_dec, H, fcin_bf, E + 3 * H, E, B, H);
        cast(ctx, 2 * H, fcin_bf, E + 3 * H, E + H, B, 2 * H);
        gemm(fcin_bf, fcw_bf, fc_b, out + (size_t)(t + 1) * B * VT, B, VT, E + 3 * H, 0);
    }
    (void)in_sizes; (void)n_in; (void)out_size; (void)ws_size;
}
